// MambaSSM_65816078844438
// MI455X (gfx1250) — matrix operand order repaired and checked by reading, not compiled, not run on hardware
//
#include <hip/hip_runtime.h>

// ---------------------------------------------------------------------------
// Mamba SSM (2 layers, L=2048, d_model=1025, d_inner=2050, dt_rank=65, N=16)
// GEMMs: v_wmma_f32_16x16x32_bf16 fed by double-buffered LDS tiles filled
// with global_load_async_to_lds_b128 (ASYNCcnt pipeline). Scan parallel over
// channels.
// ---------------------------------------------------------------------------

#define SEQ    2048
#define DM     1025        // d_model
#define DI     2050        // d_inner
#define DTR    65          // dt_rank
#define DS     16          // d_state
#define DC     4           // d_conv

// padded GEMM dims (K multiple of 32, N multiple of 128)
#define KP_IN  1056        // K pad of d_model   (in_proj)
#define NP_IN  4224        // N pad of 2*d_inner (in_proj)
#define KP_X   2080        // K pad of d_inner   (x_proj, out_proj)
#define NP_X   128         // N pad of dt_rank+2*d_state = 97
#define KP_DT  96          // K pad of dt_rank
#define NP_DT  2176        // N pad of d_inner   (dt_proj)
#define NP_OUT 1152        // N pad of d_model   (out_proj)

#define BK     32          // GEMM K step
#define LDSW   40          // LDS row stride in bf16 (80B: bank-decorrelated, 16B aligned)
#define TILE_E (128 * LDSW)   // bf16 elems per LDS tile (5120 -> 10240 B)

typedef __attribute__((ext_vector_type(16))) __bf16 v16bf;
typedef __attribute__((ext_vector_type(8)))  __bf16 v8bf;
typedef __attribute__((ext_vector_type(8)))  float  v8f;

__device__ __forceinline__ v16bf cat8(v8bf lo, v8bf hi)
{
    return __builtin_shufflevector(lo, hi, 0,1,2,3,4,5,6,7,8,9,10,11,12,13,14,15);
}

// ------------------------- bf16 WMMA GEMM ----------------------------------
// C[M,Np] = A[M,Kp] * B[Np,Kp]^T  (A,B row-major bf16, C f32 stride ldc)
// block = 8 waves -> 128x128 tile; wave -> 32x64 (2x4 WMMA frags).
// A/B tiles double-buffered in LDS via async copy engine.
__global__ __launch_bounds__(256)
void k_gemm_bf16(const __bf16* __restrict__ A, const __bf16* __restrict__ B,
                 float* __restrict__ C, int Kp, int ldc)
{
    __shared__ __align__(128) __bf16 smem[2 * 2 * TILE_E];   // [buf][A|B][128][LDSW]

    const int t    = threadIdx.x;
    const int lane = t & 31;
    const int wave = t >> 5;
    const int wn   = wave & 1;        // 2 waves along N (64 cols each)
    const int wm   = wave >> 1;       // 4 waves along M (32 rows each)

    const __bf16* Ag = A + (size_t)blockIdx.y * 128 * Kp;
    const __bf16* Bg = B + (size_t)blockIdx.x * 128 * Kp;

    const unsigned sbase = (unsigned)(size_t)&smem[0];   // LDS byte offset
    const int nk = Kp / BK;

    // per-thread async-copy coords: 512 x 16B transfers per tile, 2 per thread
    const int row0 = t >> 2;          // 0..63
    const int q0   = t & 3;           // 16B quarter within a 64B row

    auto issue = [&](int kt) {
        const int      buf  = kt & 1;
        const unsigned aoff = sbase + (unsigned)buf * (2u * TILE_E * 2u);
        const unsigned boff = aoff + TILE_E * 2u;
        const int      k0   = kt * BK;
#pragma unroll
        for (int r = 0; r < 2; ++r) {
            const int row = row0 + r * 64;
            const size_t goff = (size_t)row * Kp + k0 + q0 * 8;
            const unsigned loff = (unsigned)(row * (LDSW * 2) + q0 * 16);
            const __bf16* ga = Ag + goff;
            const __bf16* gb = Bg + goff;
            unsigned la = aoff + loff;
            unsigned lb = boff + loff;
            asm volatile("global_load_async_to_lds_b128 %0, %1, off"
                         :: "v"(la), "v"(ga) : "memory");
            asm volatile("global_load_async_to_lds_b128 %0, %1, off"
                         :: "v"(lb), "v"(gb) : "memory");
        }
    };

    v8f acc[2][4];
#pragma unroll
    for (int i = 0; i < 2; ++i)
#pragma unroll
        for (int j = 0; j < 4; ++j) acc[i][j] = (v8f){};

    const int ar = lane & 15;
    const int ak = (lane >> 4) * 8;    // A frag: lanes 16-31 hold K+8 / K+24
    const int bk = (lane >> 4) * 8;    // B frag: lanes 16-31 hold K+8 / K+24

    issue(0);

    for (int kt = 0; kt < nk; ++kt) {
        if (kt + 1 < nk) {
            issue(kt + 1);
            asm volatile("s_wait_asynccnt 0x4" ::: "memory");  // stage kt landed
        } else {
            asm volatile("s_wait_asynccnt 0x0" ::: "memory");
        }
        __syncthreads();

        const __bf16* At = smem + (kt & 1) * (2 * TILE_E);
        const __bf16* Bt = At + TILE_E;

        v16bf a[2], b[4];
#pragma unroll
        for (int i = 0; i < 2; ++i) {
            const __bf16* ap = At + (wm * 32 + i * 16 + ar) * LDSW;
            a[i] = cat8(*(const v8bf*)(ap + ak), *(const v8bf*)(ap + ak + 16));
        }
#pragma unroll
        for (int j = 0; j < 4; ++j) {
            const __bf16* bp = Bt + (wn * 64 + j * 16 + ar) * LDSW;
            b[j] = cat8(*(const v8bf*)(bp + bk), *(const v8bf*)(bp + bk + 16));
        }
#pragma unroll
        for (int i = 0; i < 2; ++i)
#pragma unroll
            for (int j = 0; j < 4; ++j)
                acc[i][j] = __builtin_amdgcn_wmma_f32_16x16x32_bf16(
                    false, a[i], false, b[j], (short)0, acc[i][j], false, false);

        __syncthreads();   // protect buf before it is refilled next iteration
    }

    // C/D layout: lanes 0-15 -> N=lane, M=e; lanes 16-31 -> N=lane-16, M=8+e
    const int crow = blockIdx.y * 128 + wm * 32 + (lane >> 4) * 8;
    const int ccol = blockIdx.x * 128 + wn * 64 + (lane & 15);
#pragma unroll
    for (int i = 0; i < 2; ++i)
#pragma unroll
        for (int j = 0; j < 4; ++j)
#pragma unroll
            for (int e = 0; e < 8; ++e)
                C[(size_t)(crow + i * 16 + e) * ldc + ccol + j * 16] = acc[i][j][e];
}

// ------------------------ elementwise kernels ------------------------------

// weight f32 (N,K) -> bf16 padded (Np,Kp), zero fill
__global__ void k_cvt_w(const float* __restrict__ src, __bf16* __restrict__ dst,
                        int N, int K, int Kp, int total)
{
    int i = blockIdx.x * blockDim.x + threadIdx.x;
    if (i >= total) return;
    int n = i / Kp, k = i - n * Kp;
    dst[i] = (n < N && k < K) ? (__bf16)src[(size_t)n * K + k] : (__bf16)0.0f;
}

// residual init: resid[l,d] = Z[d,l]   (Z is (DM, SEQ))
__global__ void k_init_resid(const float* __restrict__ Z, float* __restrict__ resid)
{
    int i = blockIdx.x * blockDim.x + threadIdx.x;
    if (i >= SEQ * DM) return;
    int l = i / DM, d = i - l * DM;
    resid[i] = Z[(size_t)d * SEQ + l];
}

__global__ void k_add_resid(float* __restrict__ resid, const float* __restrict__ outp)
{
    int i = blockIdx.x * blockDim.x + threadIdx.x;
    if (i >= SEQ * DM) return;
    int l = i / DM, d = i - l * DM;
    resid[i] += outp[(size_t)l * NP_OUT + d];
}

// LayerNorm over DM, emit bf16 padded row (KP_IN)
__global__ __launch_bounds__(256)
void k_layernorm(const float* __restrict__ X, const float* __restrict__ g,
                 const float* __restrict__ b, __bf16* __restrict__ H)
{
    __shared__ float sbuf[256], qbuf[256];
    const int l   = blockIdx.x;
    const int tid = threadIdx.x;
    const float* x = X + (size_t)l * DM;
    float s = 0.f, ss = 0.f;
    for (int d = tid; d < DM; d += 256) { float v = x[d]; s += v; ss += v * v; }
    sbuf[tid] = s; qbuf[tid] = ss;
    __syncthreads();
    for (int off = 128; off > 0; off >>= 1) {
        if (tid < off) { sbuf[tid] += sbuf[tid + off]; qbuf[tid] += qbuf[tid + off]; }
        __syncthreads();
    }
    const float mu  = sbuf[0] / (float)DM;
    const float var = qbuf[0] / (float)DM - mu * mu;
    const float r   = rsqrtf(var + 1e-5f);
    __bf16* h = H + (size_t)l * KP_IN;
    for (int d = tid; d < KP_IN; d += 256)
        h[d] = (d < DM) ? (__bf16)((x[d] - mu) * r * g[d] + b[d]) : (__bf16)0.0f;
}

// causal depthwise conv (4 taps) + SiLU ; emit f32 xs and bf16 padded x_bf
__global__ void k_conv_silu(const float* __restrict__ xz, const float* __restrict__ cw,
                            const float* __restrict__ cb, float* __restrict__ xs,
                            __bf16* __restrict__ xbf)
{
    int i = blockIdx.x * blockDim.x + threadIdx.x;
    if (i >= SEQ * KP_X) return;
    int l = i / KP_X, d = i - l * KP_X;
    if (d >= DI) { xbf[i] = (__bf16)0.0f; return; }
    float acc = cb[d];
#pragma unroll
    for (int k = 0; k < DC; ++k) {
        int t = l + k - (DC - 1);
        if (t >= 0) acc += xz[(size_t)t * NP_IN + d] * cw[d * DC + k];
    }
    float sv = acc / (1.0f + __expf(-acc));   // silu
    xs[(size_t)l * DI + d] = sv;
    xbf[i] = (__bf16)sv;
}

// dt slice (first DTR cols of x_dbl) -> bf16 padded (SEQ, KP_DT)
__global__ void k_dt_prep(const float* __restrict__ xdbl, __bf16* __restrict__ dtbf)
{
    int i = blockIdx.x * blockDim.x + threadIdx.x;
    if (i >= SEQ * KP_DT) return;
    int l = i / KP_DT, r = i - l * KP_DT;
    dtbf[i] = (r < DTR) ? (__bf16)xdbl[(size_t)l * NP_X + r] : (__bf16)0.0f;
}

// dt = softplus(dt + bias), in place on the padded (SEQ, NP_DT) buffer
__global__ void k_softplus(float* __restrict__ dtf, const float* __restrict__ dtb)
{
    int i = blockIdx.x * blockDim.x + threadIdx.x;
    if (i >= SEQ * DI) return;
    int l = i / DI, d = i - l * DI;
    float v = dtf[(size_t)l * NP_DT + d] + dtb[d];
    dtf[(size_t)l * NP_DT + d] = (v > 20.0f) ? v : log1pf(__expf(v));
}

// selective scan: one thread per channel, 16 states in registers
__global__ __launch_bounds__(256)
void k_scan(const float* __restrict__ dtf, const float* __restrict__ xdbl,
            const float* __restrict__ xs, const float* __restrict__ Alog,
            const float* __restrict__ Dsk, float* __restrict__ y)
{
    int d = blockIdx.x * blockDim.x + threadIdx.x;
    if (d >= DI) return;
    float nA[DS], h[DS];
#pragma unroll
    for (int n = 0; n < DS; ++n) {
        nA[n] = -__expf(Alog[(size_t)d * DS + n]);
        h[n]  = 0.0f;
    }
    const float dsk = Dsk[d];
    for (int t = 0; t < SEQ; ++t) {
        const float dt = dtf[(size_t)t * NP_DT + d];
        const float xv = xs[(size_t)t * DI + d];
        const float* bp = xdbl + (size_t)t * NP_X + DTR;       // B_t (16)
        const float* cp = bp + DS;                             // C_t (16)
        float acc = 0.0f;
#pragma unroll
        for (int n = 0; n < DS; ++n) {
            float dA = __expf(dt * nA[n]);
            h[n] = dA * h[n] + dt * bp[n] * xv;
            acc += h[n] * cp[n];
        }
        y[(size_t)t * DI + d] = acc + xv * dsk;
    }
}

// y = y * silu(z) -> bf16 padded (SEQ, KP_X); z = xz[:, DI:2*DI]
__global__ void k_gate(const float* __restrict__ y, const float* __restrict__ xz,
                       __bf16* __restrict__ ybf)
{
    int i = blockIdx.x * blockDim.x + threadIdx.x;
    if (i >= SEQ * KP_X) return;
    int l = i / KP_X, d = i - l * KP_X;
    if (d >= DI) { ybf[i] = (__bf16)0.0f; return; }
    float z = xz[(size_t)l * NP_IN + DI + d];
    float v = y[(size_t)l * DI + d] * (z / (1.0f + __expf(-z)));
    ybf[i] = (__bf16)v;
}

// out[d,l] = outp[l,d] + resid[l,d]
__global__ void k_final(const float* __restrict__ outp, const float* __restrict__ resid,
                        float* __restrict__ out)
{
    int i = blockIdx.x * blockDim.x + threadIdx.x;
    if (i >= SEQ * DM) return;
    int l = i / DM, d = i - l * DM;
    out[(size_t)d * SEQ + l] = outp[(size_t)l * NP_OUT + d] + resid[i];
}

// ------------------------------ launch -------------------------------------

static inline int cdiv(long a, long b) { return (int)((a + b - 1) / b); }

extern "C" void kernel_launch(void* const* d_in, const int* in_sizes, int n_in,
                              void* d_out, int out_size, void* d_ws, size_t ws_size,
                              hipStream_t stream)
{
    const float* Z    = (const float*)d_in[0];
    const float* g    = (const float*)d_in[1];
    const float* be   = (const float*)d_in[2];
    const float* wi   = (const float*)d_in[3];
    const float* cw   = (const float*)d_in[4];
    const float* cb   = (const float*)d_in[5];
    const float* wx   = (const float*)d_in[6];
    const float* wdt  = (const float*)d_in[7];
    const float* dtb  = (const float*)d_in[8];
    const float* Alog = (const float*)d_in[9];
    const float* Dsk  = (const float*)d_in[10];
    const float* wo   = (const float*)d_in[11];
    float* out = (float*)d_out;

    char* p = (char*)d_ws;
    auto alloc = [&](size_t bytes) -> void* {
        void* r = (void*)p;
        p += (bytes + 255) & ~(size_t)255;
        return r;
    };
    float*  resid  = (float*)alloc((size_t)SEQ * DM * 4);
    float*  xz     = (float*)alloc((size_t)SEQ * NP_IN * 4);
    float*  xs     = (float*)alloc((size_t)SEQ * DI * 4);
    float*  xdbl   = (float*)alloc((size_t)SEQ * NP_X * 4);
    float*  dtf    = (float*)alloc((size_t)SEQ * NP_DT * 4);
    float*  yb     = (float*)alloc((size_t)SEQ * DI * 4);
    float*  outp   = (float*)alloc((size_t)SEQ * NP_OUT * 4);
    __bf16* h_bf   = (__bf16*)alloc((size_t)SEQ * KP_IN * 2);
    __bf16* x_bf   = (__bf16*)alloc((size_t)SEQ * KP_X * 2);
    __bf16* dt_bf  = (__bf16*)alloc((size_t)SEQ * KP_DT * 2);
    __bf16* y_bf   = (__bf16*)alloc((size_t)SEQ * KP_X * 2);
    __bf16* wi_bf  = (__bf16*)alloc((size_t)NP_IN * KP_IN * 2);
    __bf16* wx_bf  = (__bf16*)alloc((size_t)NP_X * KP_X * 2);
    __bf16* wdt_bf = (__bf16*)alloc((size_t)NP_DT * KP_DT * 2);
    __bf16* wo_bf  = (__bf16*)alloc((size_t)NP_OUT * KP_X * 2);

    const int T = 256;

    // weight conversion (f32 -> padded bf16), once per call
    { int tot = NP_IN * KP_IN;
      k_cvt_w<<<cdiv(tot, T), T, 0, stream>>>(wi,  wi_bf,  2 * DI, DM,  KP_IN, tot); }
    { int tot = NP_X * KP_X;
      k_cvt_w<<<cdiv(tot, T), T, 0, stream>>>(wx,  wx_bf,  DTR + 2 * DS, DI, KP_X, tot); }
    { int tot = NP_DT * KP_DT;
      k_cvt_w<<<cdiv(tot, T), T, 0, stream>>>(wdt, wdt_bf, DI, DTR, KP_DT, tot); }
    { int tot = NP_OUT * KP_X;
      k_cvt_w<<<cdiv(tot, T), T, 0, stream>>>(wo,  wo_bf,  DM, DI, KP_X, tot); }

    k_init_resid<<<cdiv((long)SEQ * DM, T), T, 0, stream>>>(Z, resid);

    for (int layer = 0; layer < 2; ++layer) {
        if (layer > 0)
            k_add_resid<<<cdiv((long)SEQ * DM, T), T, 0, stream>>>(resid, outp);

        k_layernorm<<<SEQ, T, 0, stream>>>(resid, g, be, h_bf);

        // xz = h @ in_proj_w^T
        k_gemm_bf16<<<dim3(NP_IN / 128, SEQ / 128), T, 0, stream>>>(
            h_bf, wi_bf, xz, KP_IN, NP_IN);

        k_conv_silu<<<cdiv((long)SEQ * KP_X, T), T, 0, stream>>>(xz, cw, cb, xs, x_bf);

        // x_dbl = x @ x_proj_w^T
        k_gemm_bf16<<<dim3(NP_X / 128, SEQ / 128), T, 0, stream>>>(
            x_bf, wx_bf, xdbl, KP_X, NP_X);

        k_dt_prep<<<cdiv((long)SEQ * KP_DT, T), T, 0, stream>>>(xdbl, dt_bf);

        // dt_full = dt @ dt_proj_w^T
        k_gemm_bf16<<<dim3(NP_DT / 128, SEQ / 128), T, 0, stream>>>(
            dt_bf, wdt_bf, dtf, KP_DT, NP_DT);

        k_softplus<<<cdiv((long)SEQ * DI, T), T, 0, stream>>>(dtf, dtb);

        k_scan<<<cdiv(DI, T), T, 0, stream>>>(dtf, xdbl, xs, Alog, Dsk, yb);

        k_gate<<<cdiv((long)SEQ * KP_X, T), T, 0, stream>>>(yb, xz, y_bf);

        // out = y @ out_proj_w^T
        k_gemm_bf16<<<dim3(NP_OUT / 128, SEQ / 128), T, 0, stream>>>(
            y_bf, wo_bf, outp, KP_X, NP_OUT);
    }

    k_final<<<cdiv((long)SEQ * DM, T), T, 0, stream>>>(outp, resid, out);
}
